// SReCModel_68856915689530
// MI455X (gfx1250) — matrix operand order repaired and checked by reading, not compiled, not run on hardware
//
#include <hip/hip_runtime.h>

typedef __attribute__((ext_vector_type(16))) _Float16 v16h;
typedef __attribute__((ext_vector_type(8)))  _Float16 v8h;
typedef __attribute__((ext_vector_type(8)))  float    v8f;
typedef __attribute__((ext_vector_type(4)))  float    v4f;

#define KMIX 10
#define PATCH_LD 40   // halfs per pixel row (patch / h2 staging), 80B: 16B-aligned
#define H1_LD    72   // halfs per pixel row (h1 staging), 144B: 16B-aligned
#define RAW_LD   32   // floats per pixel row (raw staging), 128B

// Wave-local LDS ordering: staging buffers are private per wave and LDS ops of
// one wave execute in order (CDNA5 ISA), so a dscnt drain +
// compiler barrier is sufficient -- no block barrier in the hot loop.
__device__ __forceinline__ void lds_fence() {
    asm volatile("s_wait_dscnt 0x0" ::: "memory");
}

__device__ __forceinline__ v8f wmma16(v16h a, v16h b, v8f c) {
    return __builtin_amdgcn_wmma_f32_16x16x32_f16(false, a, false, b,
                                                  (short)0, c, false, false);
}

// B operand: lane gets 16 contiguous K halfs of its column from [n][ld] staging.
__device__ __forceinline__ v16h load_Bn(const _Float16* __restrict__ s, int lane,
                                        int ld, int kbase, int nbase) {
    const int n = nbase + (lane & 15);
    const int k = kbase + ((lane >> 4) ? 16 : 0);
    return *(const v16h*)(s + n * ld + k);
}

__device__ __forceinline__ int refl(int i, int n) {
    if (i < 0)  i = -i;
    if (i >= n) i = 2 * n - 2 - i;
    return i;
}

// ---------------------------------------------------------------------------
// Utility kernels
// ---------------------------------------------------------------------------
__global__ void zero_out_kernel(float* out) {
    if (threadIdx.x == 0) out[0] = 0.0f;
}

__global__ void pool_kernel(const float* __restrict__ x,
                            float* __restrict__ y,
                            float* __restrict__ xn,
                            int H, int W) {
    const int Ho = H >> 1, Wo = W >> 1;
    const int total = 3 * Ho * Wo;
    const int i = blockIdx.x * blockDim.x + threadIdx.x;
    if (i >= total) return;
    const int c   = i / (Ho * Wo);
    const int r   = (i / Wo) % Ho;
    const int col = i % Wo;
    const float* xc = x + c * H * W;
    const float v = 0.25f * (xc[(2*r    )*W + 2*col] + xc[(2*r    )*W + 2*col + 1] +
                             xc[(2*r + 1)*W + 2*col] + xc[(2*r + 1)*W + 2*col + 1]);
    y[i] = v;
    if (xn) xn[i] = fminf(fmaxf(rintf(v), 0.0f), 255.0f);
}

// ---------------------------------------------------------------------------
// Fused per-level NLL kernel (persistent blocks, grid-stride over 256-pixel
// chunks; 8 waves/block, each wave owns 32 consecutive pixels of one row).
// ---------------------------------------------------------------------------
__launch_bounds__(256)
__global__ void level_nll_kernel(const float* __restrict__ x,
                                 const float* __restrict__ y,
                                 const float* __restrict__ W1,
                                 const float* __restrict__ b1,
                                 const float* __restrict__ W2,
                                 const float* __restrict__ b2,
                                 const float* __restrict__ W3,
                                 const float* __restrict__ b3,
                                 int H, int W, int nChunks,
                                 float* __restrict__ out) {
    // 10 A tiles pre-swizzled into the WMMA A lane layout:
    // t0..3 = fused layer-1 weights (conv27 | Wsum3 | b1 | 0), t4..7 = W2(mi,ki), t8..9 = W3
    __shared__ __align__(16) _Float16 sApack[10 * 32 * 16];
    __shared__ __align__(16) unsigned char sStage[8][4608];   // per-wave reuse

    const int tid  = threadIdx.x;
    const int lane = tid & 31;
    const int wave = tid >> 5;
    const int hb   = (lane >> 4) ? 8 : 0;
    const int nlo  = lane & 15;

    // ---- pack all weight tiles once (A layout, K contiguous: k = 16*(lane/16) + j) ----
    for (int idx = tid; idx < 10 * 32 * 16; idx += 256) {
        const int t  = idx >> 9;
        const int l  = (idx >> 4) & 31;
        const int j  = idx & 15;
        const int kl = ((l >> 4) ? 16 : 0) + j;  // K within 32
        const int ml = l & 15;
        float v = 0.0f;
        if (t < 4) {                                  // layer-1 fused, K=32
            const int m = t * 16 + ml;
            if (kl < 27) {
                v = W1[m * 54 + kl];                  // W1[o][ci<3][ky][kx]
            } else if (kl < 30) {
                const float* wp = W1 + m * 54 + (3 + (kl - 27)) * 9;  // Wsum
                v = wp[0]+wp[1]+wp[2]+wp[3]+wp[4]+wp[5]+wp[6]+wp[7]+wp[8];
            } else if (kl == 30) {
                v = b1[m];                            // bias via constant-1 input
            }
        } else if (t < 8) {                           // W2 tile (mi,ki)
            const int mi = (t - 4) >> 1, ki = (t - 4) & 1;
            v = W2[(mi * 16 + ml) * 64 + ki * 32 + kl];
        } else {                                      // W3 tile, rows>=30 zero
            const int m = (t - 8) * 16 + ml;
            if (m < 30) v = W3[m * 32 + kl];
        }
        sApack[idx] = (_Float16)v;
    }
    __syncthreads();

    // ---- A operands resident in registers for the whole kernel ----
    v16h Awa[4], Aw2[4], Aw3[2];
#pragma unroll
    for (int t = 0; t < 4; ++t) Awa[t] = *(const v16h*)&sApack[(t * 32 + lane) * 16];
#pragma unroll
    for (int t = 0; t < 4; ++t) Aw2[t] = *(const v16h*)&sApack[((4 + t) * 32 + lane) * 16];
#pragma unroll
    for (int t = 0; t < 2; ++t) Aw3[t] = *(const v16h*)&sApack[((8 + t) * 32 + lane) * 16];

    // layer-2/3 bias values this lane needs (C/D rows it owns)
    float b2r[16], b3r[16];
#pragma unroll
    for (int mi = 0; mi < 2; ++mi)
#pragma unroll
        for (int r = 0; r < 8; ++r) {
            const int m = mi * 16 + hb + r;
            b2r[mi * 8 + r] = b2[m];
            b3r[mi * 8 + r] = (m < 30) ? b3[m] : 0.0f;
        }

    _Float16* stageH = (_Float16*)sStage[wave];
    float*    stageF = (float*)sStage[wave];
    const int HW = H * W, Wo = W >> 1, Ho = H >> 1;

    float nll_acc = 0.0f;

    for (int chunk = blockIdx.x; chunk < nChunks; chunk += gridDim.x) {
        const int p   = chunk * 256 + tid;
        const int row = p / W;
        const int col = p % W;

        // ---- build per-pixel patch vector (pixel-major staging) ----
        v8h pk[4] = {};
        float xc_center[3];
#pragma unroll
        for (int ci = 0; ci < 3; ++ci) {
            const float* xp = x + ci * HW;
#pragma unroll
            for (int ky = 0; ky < 3; ++ky) {
                const int rr = refl(row + ky - 1, H);
#pragma unroll
                for (int kx = 0; kx < 3; ++kx) {
                    const int cc = refl(col + kx - 1, W);
                    const float v = xp[rr * W + cc];
                    const int k = ci * 9 + ky * 3 + kx;
                    pk[k >> 3][k & 7] = (_Float16)v;
                    if (ky == 1 && kx == 1) xc_center[ci] = v;
                }
            }
            const float yv = y[ci * Ho * Wo + (row >> 1) * Wo + (col >> 1)];
            const int k = 27 + ci;
            pk[k >> 3][k & 7] = (_Float16)yv;
        }
        pk[3][6] = (_Float16)1.0f;   // K=30: constant-1 -> b1
        pk[3][7] = (_Float16)0.0f;   // K=31: pad
        {
            v8h* prow = (v8h*)(stageH + lane * PATCH_LD);
            prow[0] = pk[0]; prow[1] = pk[1]; prow[2] = pk[2]; prow[3] = pk[3];
        }
        lds_fence();

        // ---- layer 1: h1 = relu(WA @ patch), 64x32, 8 WMMAs ----
        const v16h B0 = load_Bn(stageH, lane, PATCH_LD, 0, 0);
        const v16h B1 = load_Bn(stageH, lane, PATCH_LD, 0, 16);
        v8f h1t[4][2];
#pragma unroll
        for (int mi = 0; mi < 4; ++mi) {
            v8f c0 = {}, c1 = {};
            h1t[mi][0] = wmma16(Awa[mi], B0, c0);
            h1t[mi][1] = wmma16(Awa[mi], B1, c1);
        }
        lds_fence();
#pragma unroll
        for (int mi = 0; mi < 4; ++mi)
#pragma unroll
            for (int ni = 0; ni < 2; ++ni) {
                v8h hv;
#pragma unroll
                for (int r = 0; r < 8; ++r)
                    hv[r] = (_Float16)fmaxf(h1t[mi][ni][r], 0.0f);
                *(v8h*)(stageH + (ni * 16 + nlo) * H1_LD + mi * 16 + hb) = hv;
            }
        lds_fence();

        // ---- layer 2: h2 = relu(W2 @ h1 + b2), 32x32, 8 WMMAs ----
        v16h Bh[2][2];
#pragma unroll
        for (int ki = 0; ki < 2; ++ki)
#pragma unroll
            for (int ni = 0; ni < 2; ++ni)
                Bh[ki][ni] = load_Bn(stageH, lane, H1_LD, ki * 32, ni * 16);
        v8f h2t[2][2];
#pragma unroll
        for (int mi = 0; mi < 2; ++mi)
#pragma unroll
            for (int ni = 0; ni < 2; ++ni) {
                v8f c = {};
                c = wmma16(Aw2[mi * 2 + 0], Bh[0][ni], c);
                c = wmma16(Aw2[mi * 2 + 1], Bh[1][ni], c);
                h2t[mi][ni] = c;
            }
        lds_fence();
#pragma unroll
        for (int mi = 0; mi < 2; ++mi)
#pragma unroll
            for (int ni = 0; ni < 2; ++ni) {
                v8h hv;
#pragma unroll
                for (int r = 0; r < 8; ++r)
                    hv[r] = (_Float16)fmaxf(h2t[mi][ni][r] + b2r[mi * 8 + r], 0.0f);
                *(v8h*)(stageH + (ni * 16 + nlo) * PATCH_LD + mi * 16 + hb) = hv;
            }
        lds_fence();

        // ---- layer 3: raw = W3 @ h2 + b3, 30x32 (padded 32), 4 WMMAs ----
        v16h Bg[2];
#pragma unroll
        for (int ni = 0; ni < 2; ++ni)
            Bg[ni] = load_Bn(stageH, lane, PATCH_LD, 0, ni * 16);
        v8f rt[2][2];
#pragma unroll
        for (int mi = 0; mi < 2; ++mi)
#pragma unroll
            for (int ni = 0; ni < 2; ++ni) {
                v8f c = {};
                rt[mi][ni] = wmma16(Aw3[mi], Bg[ni], c);
            }
        lds_fence();
#pragma unroll
        for (int mi = 0; mi < 2; ++mi)
#pragma unroll
            for (int ni = 0; ni < 2; ++ni) {
                v4f r0, r1;
#pragma unroll
                for (int r = 0; r < 4; ++r) {
                    r0[r] = rt[mi][ni][r]     + b3r[mi * 8 + r];
                    r1[r] = rt[mi][ni][4 + r] + b3r[mi * 8 + 4 + r];
                }
                float* dst = stageF + (ni * 16 + nlo) * RAW_LD + mi * 16 + hb;
                *(v4f*)dst       = r0;
                *(v4f*)(dst + 4) = r1;
            }
        lds_fence();

        // ---- discretized logistic mixture NLL; lane <-> its pixel row ----
        float rv[32];
        {
            const float* rrow = stageF + lane * RAW_LD;
#pragma unroll
            for (int q = 0; q < 8; ++q) {
                const v4f tv = *(const v4f*)(rrow + q * 4);
                rv[q*4+0] = tv[0]; rv[q*4+1] = tv[1];
                rv[q*4+2] = tv[2]; rv[q*4+3] = tv[3];
            }
        }
        float lw[KMIX], mu[KMIX], sv[KMIX];
        float wmax = -1e30f;
#pragma unroll
        for (int k = 0; k < KMIX; ++k) {
            lw[k] = rv[3 * k + 0];
            mu[k] = rv[3 * k + 1];
            sv[k] = fminf(fmaxf(__expf(rv[3 * k + 2]), 1e-5f), 1e5f);
            wmax  = fmaxf(wmax, lw[k]);
        }
        float wsum = 0.0f;
#pragma unroll
        for (int k = 0; k < KMIX; ++k) { lw[k] = __expf(lw[k] - wmax); wsum += lw[k]; }
        const float inv_ws = 1.0f / wsum;

        float nll = 0.0f;
#pragma unroll
        for (int c = 0; c < 3; ++c) {
            const float t = xc_center[c];
            float mix = 0.0f;
#pragma unroll
            for (int k = 0; k < KMIX; ++k) {
                const float is = 1.0f / sv[k];
                const float d  = t - mu[k];
                const float sa = 1.0f / (1.0f + __expf(-(d + 0.5f) * is));
                const float sb = 1.0f / (1.0f + __expf(-(d - 0.5f) * is));
                mix += lw[k] * (sa - sb);
            }
            mix = fmaxf(mix * inv_ws, 1e-10f);
            nll -= __logf(mix);
        }
        nll_acc += nll;
        lds_fence();   // mixture reads complete before next iteration's stores
    }

    // wave32 reduce, single atomic per wave
#pragma unroll
    for (int off = 16; off > 0; off >>= 1) nll_acc += __shfl_down(nll_acc, off, 32);
    if (lane == 0) atomicAdd(out, nll_acc);
}

// ---------------------------------------------------------------------------
// Launcher
// ---------------------------------------------------------------------------
extern "C" void kernel_launch(void* const* d_in, const int* in_sizes, int n_in,
                              void* d_out, int out_size, void* d_ws, size_t ws_size,
                              hipStream_t stream) {
    (void)in_sizes; (void)n_in; (void)out_size; (void)ws_size;

    const float* image = (const float*)d_in[0];
    float* ws = (float*)d_ws;
    float* y0 = ws;                       // 3*256*256
    float* x1 = y0 + 3 * 256 * 256;       // 3*256*256
    float* y1 = x1 + 3 * 256 * 256;       // 3*128*128
    float* x2 = y1 + 3 * 128 * 128;       // 3*128*128
    float* y2 = x2 + 3 * 128 * 128;       // 3*64*64
    float* out = (float*)d_out;

    zero_out_kernel<<<1, 32, 0, stream>>>(out);

    { const int t = 3 * 256 * 256;
      pool_kernel<<<(t + 255) / 256, 256, 0, stream>>>(image, y0, x1, 512, 512); }
    { const int t = 3 * 128 * 128;
      pool_kernel<<<(t + 255) / 256, 256, 0, stream>>>(x1, y1, x2, 256, 256); }
    { const int t = 3 * 64 * 64;
      pool_kernel<<<(t + 255) / 256, 256, 0, stream>>>(x2, y2, nullptr, 128, 128); }

    const float* xs[3]  = { image, x1, x2 };
    const float* ysl[3] = { y0, y1, y2 };
    const int    Hs[3]  = { 512, 256, 128 };
    for (int l = 0; l < 3; ++l) {
        const float* W1 = (const float*)d_in[1 + 6 * l + 0];
        const float* b1 = (const float*)d_in[1 + 6 * l + 1];
        const float* W2 = (const float*)d_in[1 + 6 * l + 2];
        const float* b2 = (const float*)d_in[1 + 6 * l + 3];
        const float* W3 = (const float*)d_in[1 + 6 * l + 4];
        const float* b3 = (const float*)d_in[1 + 6 * l + 5];
        const int H = Hs[l], W = Hs[l];
        const int nChunks = (H * W) / 256;           // exact at every level
        const int blocks  = nChunks < 256 ? nChunks : 256;
        level_nll_kernel<<<blocks, 256, 0, stream>>>(
            xs[l], ysl[l], W1, b1, W2, b2, W3, b3, H, W, nChunks, out);
    }
}
